// AMPNetClassifier_55018531062156
// MI455X (gfx1250) — compile-verified, matrix operand order checked by reading, not run on hardware
//
#include <hip/hip_runtime.h>
#include <cstdint>

typedef __attribute__((ext_vector_type(16))) _Float16 v16h;
typedef __attribute__((ext_vector_type(8)))  float    v8f;

#define NF    8      // tokens per node
#define DIM   16     // embed dim
#define FD    128    // NF*DIM
#define OUTC  7

// ---------------------------------------------------------------------------
// helpers
// ---------------------------------------------------------------------------

// Build a 16-bit A/B WMMA fragment (16x16x32 shape, K=16 real, K=16..31 zero).
// Caller passes the pointer already offset by this lane's K-half (0 or 8).
// Lane layout (ISA 7.12.2): lanes 0-15 hold K0..7 in v[0..3], lanes 16-31 hold
// K8..15 in v[0..3]; v[4..7] (K16..31) are zero.
static __device__ __forceinline__ v16h make_frag(const float* p) {
  const float4* q = reinterpret_cast<const float4*>(p);
  float4 a = q[0], b = q[1];
  v16h f = {};
  f[0] = (_Float16)a.x; f[1] = (_Float16)a.y; f[2] = (_Float16)a.z; f[3] = (_Float16)a.w;
  f[4] = (_Float16)b.x; f[5] = (_Float16)b.y; f[6] = (_Float16)b.z; f[7] = (_Float16)b.w;
  return f;
}

static __device__ __forceinline__ v8f wmma_fp16(v16h a, v16h b, v8f c) {
  return __builtin_amdgcn_wmma_f32_16x16x32_f16(
      /*neg_a=*/false, a, /*neg_b=*/false, b,
      /*c_mod=*/(short)0, c, /*reuse_a=*/false, /*reuse_b=*/false);
}

// Wave-internal LDS producer->consumer sync (LDS region is wave-private, so a
// full workgroup barrier is unnecessary; DS ops are in-order per wave, we just
// need the counter drained and the scheduler fenced).
static __device__ __forceinline__ void wave_lds_sync() {
  __builtin_amdgcn_wave_barrier();
  asm volatile("s_wait_dscnt 0" ::: "memory");
  __builtin_amdgcn_wave_barrier();
}

static __device__ __forceinline__ void atom_add_f32(float* p, float v) {
  // lowers to global_atomic_add_f32 (no CAS loop)
  unsafeAtomicAdd(p, v);
}

// ---------------------------------------------------------------------------
// fused per-edge multi-head attention + scatter-add (one AMP conv layer)
// 2 edges per 16x16 WMMA tile; one wave per tile iteration.
// ---------------------------------------------------------------------------
__global__ __launch_bounds__(256) void edge_attn_kernel(
    const float* __restrict__ x, const int* __restrict__ srcIdx,
    const int* __restrict__ dstIdx, const float* __restrict__ wqkv,
    const float* __restrict__ bqkv, const float* __restrict__ wo,
    const float* __restrict__ bo, float* __restrict__ accum,
    int E, int ntiles)
{
  __shared__ float lds[8][4 * 256];           // 8 waves x (Q,K,V,O) 16x16 f32 tiles
  const int lane = threadIdx.x & 31;
  const int wave = threadIdx.x >> 5;
  float* Qs = &lds[wave][0];
  float* Ks = &lds[wave][256];
  float* Vs = &lds[wave][512];
  float* Os = &lds[wave][768];

  const int c  = lane & 15;                   // column / row index (mod 16)
  const int hi = lane >> 4;                   // lane group
  const int kh = hi * 8;                      // K-half offset for A/B frags
  const int rh = hi * 8;                      // row-half offset for C/D tiles

  // Weight B-fragments: B[k][n] = W[n][k] -> lane (col n=c) holds row n of W.
  const v16h fbq = make_frag(wqkv + (size_t)(c)      * DIM + kh);
  const v16h fbk = make_frag(wqkv + (size_t)(16 + c) * DIM + kh);
  const v16h fbv = make_frag(wqkv + (size_t)(32 + c) * DIM + kh);
  const v16h fbo = make_frag(wo   + (size_t)c        * DIM + kh);
  const float biq = bqkv[c], bik = bqkv[16 + c], biv = bqkv[32 + c], bio = bo[c];

  const int tok  = c & 7;                     // token within edge for A rows
  const int esel = c >> 3;                    // which edge of the pair (A rows)

  const int gw = blockIdx.x * 8 + wave;
  const int nw = gridDim.x * 8;
  for (int t = gw; t < ntiles; t += nw) {
    const int e0 = 2 * t;
    const bool has2 = (e0 + 1 < E);
    const int e1 = has2 ? e0 + 1 : e0;
    const int eA = esel ? e1 : e0;
    const int ndst = dstIdx[eA];
    const int nsrc = srcIdx[eA];

    // prefetch next tile's gather rows into cache (global_prefetch_b8)
    const int t2 = t + nw;
    if (t2 < ntiles) {
      const int p0 = 2 * t2;
      const int pe = esel ? ((p0 + 1 < E) ? p0 + 1 : p0) : p0;
      __builtin_prefetch(x + (size_t)dstIdx[pe] * FD + tok * DIM, 0, 0);
      __builtin_prefetch(x + (size_t)srcIdx[pe] * FD + tok * DIM, 0, 0);
    }

    // A tiles: rows = 2 edges x 8 tokens; query from dst, key/value from src
    const v16h ai = make_frag(x + (size_t)ndst * FD + tok * DIM + kh);
    const v16h aj = make_frag(x + (size_t)nsrc * FD + tok * DIM + kh);

    v8f zero = {};
    v8f qa = wmma_fp16(ai, fbq, zero);
    v8f ka = wmma_fp16(aj, fbk, zero);
    v8f va = wmma_fp16(aj, fbv, zero);

    // stage Q/K/V (+bias) to wave-private LDS, [row][col] f32
#pragma unroll
    for (int r = 0; r < 8; ++r) {
      Qs[(rh + r) * 16 + c] = qa[r] + biq;
      Ks[(rh + r) * 16 + c] = ka[r] + bik;
      Vs[(rh + r) * 16 + c] = va[r] + biv;
    }
    wave_lds_sync();

    // attention core: 64 (row,head) tasks per tile, 2 per lane.
    // scores[f,g] = 0.5 * sum_d Q[f,4h+d] K[g,4h+d]; softmax over g (8 tokens)
#pragma unroll
    for (int half = 0; half < 2; ++half) {
      const int task = lane + 32 * half;
      const int f = task >> 2;                // tile row (edge*8+token)
      const int h = task & 3;                 // head
      const int gb = (f >> 3) << 3;           // token-row base of this edge
      const float4 qv = *reinterpret_cast<const float4*>(Qs + f * 16 + h * 4);
      float s[8];
      float mx = -3.0e38f;
#pragma unroll
      for (int g = 0; g < 8; ++g) {
        const float4 kv = *reinterpret_cast<const float4*>(Ks + (gb + g) * 16 + h * 4);
        s[g] = 0.5f * (qv.x * kv.x + qv.y * kv.y + qv.z * kv.z + qv.w * kv.w);
        mx = fmaxf(mx, s[g]);
      }
      float sum = 0.0f;
#pragma unroll
      for (int g = 0; g < 8; ++g) { s[g] = __expf(s[g] - mx); sum += s[g]; }
      const float inv = 1.0f / sum;
      float o0 = 0.f, o1 = 0.f, o2 = 0.f, o3 = 0.f;
#pragma unroll
      for (int g = 0; g < 8; ++g) {
        const float4 vv = *reinterpret_cast<const float4*>(Vs + (gb + g) * 16 + h * 4);
        const float p = s[g] * inv;
        o0 += p * vv.x; o1 += p * vv.y; o2 += p * vv.z; o3 += p * vv.w;
      }
      float* orow = Os + f * 16 + h * 4;
      orow[0] = o0; orow[1] = o1; orow[2] = o2; orow[3] = o3;
    }
    wave_lds_sync();

    // output projection: attn_out @ wo^T, then scatter-add msg rows to dst
    const v16h ao = make_frag(Os + c * 16 + kh);
    v8f oa = wmma_fp16(ao, fbo, zero);

    const int edl = hi ? e1 : e0;             // C/D row-half selects edge
    if (hi == 0 || has2) {
      float* dstrow = accum + (size_t)dstIdx[edl] * FD + c;
#pragma unroll
      for (int r = 0; r < 8; ++r) {
        atom_add_f32(dstrow + r * DIM, oa[r] + bio);
      }
    }
  }
}

// ---------------------------------------------------------------------------
// elementwise / epilogue kernels
// ---------------------------------------------------------------------------
__global__ void zero_kernel(float4* __restrict__ p, long long n4) {
  long long i = (long long)blockIdx.x * blockDim.x + threadIdx.x;
  const long long stride = (long long)gridDim.x * blockDim.x;
  const float4 z = make_float4(0.f, 0.f, 0.f, 0.f);
  for (; i < n4; i += stride) p[i] = z;
}

__global__ void elu_kernel(float* __restrict__ p, long long n) {
  long long i = (long long)blockIdx.x * blockDim.x + threadIdx.x;
  const long long stride = (long long)gridDim.x * blockDim.x;
  for (; i < n; i += stride) {
    const float v = p[i];
    p[i] = (v > 0.0f) ? v : (__expf(v) - 1.0f);
  }
}

__global__ void head_kernel(const float* __restrict__ h, const float* __restrict__ w,
                            const float* __restrict__ b, float* __restrict__ out, int N) {
  const int n = blockIdx.x * blockDim.x + threadIdx.x;
  if (n >= N) return;
  const float4* row = reinterpret_cast<const float4*>(h + (size_t)n * FD);
  float l[OUTC];
#pragma unroll
  for (int o = 0; o < OUTC; ++o) {
    const float4* wr = reinterpret_cast<const float4*>(w + (size_t)o * FD);
    float acc = b[o];
#pragma unroll
    for (int k = 0; k < FD / 4; ++k) {
      const float4 xv = row[k];
      const float4 wv = wr[k];
      acc += xv.x * wv.x + xv.y * wv.y + xv.z * wv.z + xv.w * wv.w;
    }
    l[o] = acc;
  }
  float mx = l[0];
#pragma unroll
  for (int o = 1; o < OUTC; ++o) mx = fmaxf(mx, l[o]);
  float sum = 0.0f;
#pragma unroll
  for (int o = 0; o < OUTC; ++o) sum += __expf(l[o] - mx);
  const float lse = __logf(sum);
#pragma unroll
  for (int o = 0; o < OUTC; ++o) out[(size_t)n * OUTC + o] = l[o] - mx - lse;
}

// ---------------------------------------------------------------------------
// launch
// ---------------------------------------------------------------------------
extern "C" void kernel_launch(void* const* d_in, const int* in_sizes, int n_in,
                              void* d_out, int out_size, void* d_ws, size_t ws_size,
                              hipStream_t stream) {
  (void)n_in; (void)out_size; (void)ws_size;
  const float* x     = (const float*)d_in[0];
  const int*   ei    = (const int*)  d_in[1];
  const float* w1qkv = (const float*)d_in[2];
  const float* b1qkv = (const float*)d_in[3];
  const float* w1o   = (const float*)d_in[4];
  const float* b1o   = (const float*)d_in[5];
  const float* w2qkv = (const float*)d_in[6];
  const float* b2qkv = (const float*)d_in[7];
  const float* w2o   = (const float*)d_in[8];
  const float* b2o   = (const float*)d_in[9];
  const float* outw  = (const float*)d_in[10];
  const float* outb  = (const float*)d_in[11];

  const int N = in_sizes[0] / FD;
  const int E = in_sizes[1] / 2;
  const int* src = ei;
  const int* dst = ei + E;

  float* h1 = (float*)d_ws;
  float* h2 = h1 + (size_t)N * FD;

  // zero both accumulators (atomics accumulate into them every call)
  const long long n4 = (2LL * N * FD) / 4;
  zero_kernel<<<256, 256, 0, stream>>>(reinterpret_cast<float4*>(h1), n4);

  const int ntiles = (E + 1) / 2;
  const int nblocks = 640;   // 640 blocks x 8 waves = 5120 waves, ~31 tiles each

  edge_attn_kernel<<<nblocks, 256, 0, stream>>>(x, src, dst, w1qkv, b1qkv, w1o, b1o,
                                                h1, E, ntiles);
  elu_kernel<<<1024, 256, 0, stream>>>(h1, (long long)N * FD);
  edge_attn_kernel<<<nblocks, 256, 0, stream>>>(h1, src, dst, w2qkv, b2qkv, w2o, b2o,
                                                h2, E, ntiles);
  elu_kernel<<<1024, 256, 0, stream>>>(h2, (long long)N * FD);
  head_kernel<<<(N + 255) / 256, 256, 0, stream>>>(h2, outw, outb, (float*)d_out, N);
}
